// SelfAttention_24008867184860
// MI455X (gfx1250) — matrix operand order repaired and checked by reading, not compiled, not run on hardware
//
#include <hip/hip_runtime.h>
#include <math.h>

typedef __attribute__((ext_vector_type(16))) _Float16 v16h;
typedef __attribute__((ext_vector_type(8)))  _Float16 v8h;
typedef __attribute__((ext_vector_type(8)))  float    v8f;

#define SEQ  2048
#define NH   16
#define HD   64
#define QT   16      // query rows per wave
#define MT   128     // query rows per workgroup (8 waves)
#define NT   32      // keys per iteration
#define KSTR 72      // LDS stride (halves) for K tile rows (conflict-free, 144B rows 16B-aligned)
#define VSTR 40      // LDS stride (halves) for V^T rows
#define PSTR 40      // LDS stride (halves) for P staging rows

__device__ __forceinline__ v16h join16(v8h lo, v8h hi) {
  return __builtin_shufflevector(lo, hi, 0,1,2,3,4,5,6,7,8,9,10,11,12,13,14,15);
}

// DPP16 lane permute (pure VALU, no LDS). CTRL: 0xB1=quad_perm xor1, 0x4E=quad_perm xor2,
// 0x141=row_half_mirror, 0x140=row_mirror.
template <int CTRL>
__device__ __forceinline__ float dpp_mov(float x) {
  return __int_as_float(
      __builtin_amdgcn_update_dpp(0, __float_as_int(x), CTRL, 0xf, 0xf, true));
}

// reduce across each 16-lane half; after xor1+xor2 quads are homogeneous, so the
// half-row mirror acts as xor4 and the row mirror acts as xor8.
__device__ __forceinline__ float rowmax16(float x) {
  x = fmaxf(x, dpp_mov<0xB1>(x));
  x = fmaxf(x, dpp_mov<0x4E>(x));
  x = fmaxf(x, dpp_mov<0x141>(x));
  x = fmaxf(x, dpp_mov<0x140>(x));
  return x;
}
__device__ __forceinline__ float rowsum16(float x) {
  x += dpp_mov<0xB1>(x);
  x += dpp_mov<0x4E>(x);
  x += dpp_mov<0x141>(x);
  x += dpp_mov<0x140>(x);
  return x;
}

__global__ __launch_bounds__(256)
void fa_fwd_f16wmma(const float* __restrict__ qkv, float* __restrict__ out) {
  const int qblk = blockIdx.x, h = blockIdx.y, b = blockIdx.z;
  const int tid  = threadIdx.x;
  const int wave = tid >> 5;
  const int lane = tid & 31;
  const int l16  = lane & 15;
  const int hv   = lane >> 4;       // lane half: 0 or 1

  // double-buffered K (row-major key x dim) and V^T (dim x key); P staging is wave-private
  __shared__ __align__(16) _Float16 Ks[2][NT * KSTR];
  __shared__ __align__(16) _Float16 Vt[2][HD * VSTR];
  __shared__ __align__(16) _Float16 Ps[8 * QT * PSTR];

  const int qbase = qblk * MT + wave * QT;
  const int qmax  = qbase + QT - 1;

  const float QSCALE = 0.125f * 1.44269504088896340736f;   // 1/sqrt(64) * log2(e)
  const float NEGL2  = -10000.0f * 1.44269504088896340736f;

  // ---------------- Q: 16x64 fp32 rows -> two 16x32 f16 A-fragments (scale folded) ----------------
  v16h qa0, qa1;
  {
    const int qrow = qbase + l16;
    const float* qp = qkv + ((size_t)(b * SEQ + qrow) * 3 + 0) * (NH * HD) + h * HD;
    float tmp[8];
    #pragma unroll
    for (int c = 0; c < 4; ++c) {                // c=0,1 -> qa0 ; c=2,3 -> qa1
      const int dbase = c * 16 + hv * 8;
      const float4 f0 = *reinterpret_cast<const float4*>(qp + dbase);
      const float4 f1 = *reinterpret_cast<const float4*>(qp + dbase + 4);
      tmp[0]=f0.x; tmp[1]=f0.y; tmp[2]=f0.z; tmp[3]=f0.w;
      tmp[4]=f1.x; tmp[5]=f1.y; tmp[6]=f1.z; tmp[7]=f1.w;
      #pragma unroll
      for (int j = 0; j < 8; ++j) {
        const _Float16 e = (_Float16)(tmp[j] * QSCALE);
        if (c < 2) qa0[(c & 1) * 8 + j] = e;
        else       qa1[(c & 1) * 8 + j] = e;
      }
    }
  }

  // per-thread piece of the cooperative K/V tile load (2048 f32 per tensor, 256 threads)
  const int e0 = tid * 4,         row0 = e0 >> 6, d0 = e0 & 63;
  const int e1 = (256 + tid) * 4, row1 = e1 >> 6, d1 = e1 & 63;
  float4 kr0, kr1, vr0, vr1;

  auto gloadKV = [&](int kbase) {
    const size_t i0 = ((size_t)(b * SEQ + kbase + row0) * 3 + 1) * (NH * HD) + h * HD + d0;
    kr0 = *reinterpret_cast<const float4*>(qkv + i0);
    vr0 = *reinterpret_cast<const float4*>(qkv + i0 + NH * HD);
    const size_t i1 = ((size_t)(b * SEQ + kbase + row1) * 3 + 1) * (NH * HD) + h * HD + d1;
    kr1 = *reinterpret_cast<const float4*>(qkv + i1);
    vr1 = *reinterpret_cast<const float4*>(qkv + i1 + NH * HD);
  };
  auto lstoreKV = [&](int buf) {
    _Float16* kp = &Ks[buf][row0 * KSTR + d0];
    kp[0]=(_Float16)kr0.x; kp[1]=(_Float16)kr0.y; kp[2]=(_Float16)kr0.z; kp[3]=(_Float16)kr0.w;
    kp = &Ks[buf][row1 * KSTR + d1];
    kp[0]=(_Float16)kr1.x; kp[1]=(_Float16)kr1.y; kp[2]=(_Float16)kr1.z; kp[3]=(_Float16)kr1.w;
    Vt[buf][(d0+0)*VSTR + row0]=(_Float16)vr0.x; Vt[buf][(d0+1)*VSTR + row0]=(_Float16)vr0.y;
    Vt[buf][(d0+2)*VSTR + row0]=(_Float16)vr0.z; Vt[buf][(d0+3)*VSTR + row0]=(_Float16)vr0.w;
    Vt[buf][(d1+0)*VSTR + row1]=(_Float16)vr1.x; Vt[buf][(d1+1)*VSTR + row1]=(_Float16)vr1.y;
    Vt[buf][(d1+2)*VSTR + row1]=(_Float16)vr1.z; Vt[buf][(d1+3)*VSTR + row1]=(_Float16)vr1.w;
  };

  float m_i[8], l_i[8];
  v8f o0 = {}, o1 = {}, o2 = {}, o3 = {};
  #pragma unroll
  for (int r = 0; r < 8; ++r) { m_i[r] = -3.0e38f; l_i[r] = 0.0f; }

  const int njb = qblk * 4 + 4;   // key blocks covering this Q tile's causal range

  // prologue: fill buffer 0
  gloadKV(0);
  lstoreKV(0);
  __syncthreads();

  for (int jb = 0; jb < njb; ++jb) {
    const int kbase = jb * NT;
    const int buf   = jb & 1;

    // prefetch next K/V tile into registers (global loads overlap with the WMMAs below)
    if (jb + 1 < njb) gloadKV(kbase + NT);

    if (kbase <= qmax) {          // wave-uniform: skip fully-masked key blocks
      // B-fragment of K^T (32k x 16n): lane = key row n_off+l16; elems = dims k_off + hv*8 .. +7, k_off + 16 + hv*8 .. +7
      auto ldK = [&](int n_off, int k_off) -> v16h {
        const _Float16* p = &Ks[buf][(n_off + l16) * KSTR + k_off + hv * 8];
        return join16(*reinterpret_cast<const v8h*>(p),
                      *reinterpret_cast<const v8h*>(p + 16));
      };
      v8f s0 = {}, s1 = {};
      s0 = __builtin_amdgcn_wmma_f32_16x16x32_f16(false, qa0, false, ldK(0,  0), (short)0, s0, false, false);
      s0 = __builtin_amdgcn_wmma_f32_16x16x32_f16(false, qa1, false, ldK(0, 32), (short)0, s0, false, false);
      s1 = __builtin_amdgcn_wmma_f32_16x16x32_f16(false, qa0, false, ldK(16, 0), (short)0, s1, false, false);
      s1 = __builtin_amdgcn_wmma_f32_16x16x32_f16(false, qa1, false, ldK(16,32), (short)0, s1, false, false);

      // causal mask (only blocks crossing the diagonal); scores are in log2 domain
      if (kbase + NT - 1 > qbase) {
        #pragma unroll
        for (int r = 0; r < 8; ++r) {
          const int qi = qbase + r + 8 * hv;           // C-layout row for (VGPR r, lane half hv)
          if (kbase + l16 > qi)      s0[r] += NEGL2;
          if (kbase + 16 + l16 > qi) s1[r] += NEGL2;
        }
      }

      // online softmax: DPP reductions within each 16-lane half (no LDS traffic)
      _Float16* pw = &Ps[wave * QT * PSTR];
      #pragma unroll
      for (int r = 0; r < 8; ++r) {
        const float rm    = rowmax16(fmaxf(s0[r], s1[r]));
        const float mn    = fmaxf(m_i[r], rm);
        const float alpha = __builtin_amdgcn_exp2f(m_i[r] - mn);
        m_i[r] = mn;
        const float p0 = __builtin_amdgcn_exp2f(s0[r] - mn);
        const float p1 = __builtin_amdgcn_exp2f(s1[r] - mn);
        l_i[r] = l_i[r] * alpha + rowsum16(p0 + p1);
        o0[r] *= alpha; o1[r] *= alpha; o2[r] *= alpha; o3[r] *= alpha;
        // stage P (C-layout element (row r+8*hv, col n)) into wave-private LDS
        pw[(r + 8 * hv) * PSTR + l16]      = (_Float16)p0;
        pw[(r + 8 * hv) * PSTR + 16 + l16] = (_Float16)p1;
      }

      // reload P as an A-fragment (16x32): lane = row l16, key chunks hv*8 and 16+hv*8
      const _Float16* pr = &Ps[wave * QT * PSTR + l16 * PSTR];
      const v16h pa = join16(*reinterpret_cast<const v8h*>(pr + hv * 8),
                             *reinterpret_cast<const v8h*>(pr + 16 + hv * 8));

      // B-fragment of V (32k x 16n): lane = dim row d_off+l16; elems = keys hv*8 .. +7, 16 + hv*8 .. +7
      auto ldV = [&](int d_off) -> v16h {
        const _Float16* p = &Vt[buf][(d_off + l16) * VSTR + hv * 8];
        return join16(*reinterpret_cast<const v8h*>(p),
                      *reinterpret_cast<const v8h*>(p + 16));
      };
      o0 = __builtin_amdgcn_wmma_f32_16x16x32_f16(false, pa, false, ldV(0),  (short)0, o0, false, false);
      o1 = __builtin_amdgcn_wmma_f32_16x16x32_f16(false, pa, false, ldV(16), (short)0, o1, false, false);
      o2 = __builtin_amdgcn_wmma_f32_16x16x32_f16(false, pa, false, ldV(32), (short)0, o2, false, false);
      o3 = __builtin_amdgcn_wmma_f32_16x16x32_f16(false, pa, false, ldV(48), (short)0, o3, false, false);
    }

    // write the prefetched tile into the other buffer; one barrier per iteration
    if (jb + 1 < njb) lstoreKV(buf ^ 1);
    __syncthreads();
  }

  // ---------------- epilogue: normalize and store fp32 output (b, t, h, d) ----------------
  #pragma unroll
  for (int r = 0; r < 8; ++r) {
    const float inv = 1.0f / l_i[r];
    const int t = qbase + r + 8 * hv;
    float* op = out + ((size_t)(b * SEQ + t) * NH + h) * HD;
    op[0  + l16] = o0[r] * inv;
    op[16 + l16] = o1[r] * inv;
    op[32 + l16] = o2[r] * inv;
    op[48 + l16] = o3[r] * inv;
  }
}

extern "C" void kernel_launch(void* const* d_in, const int* in_sizes, int n_in,
                              void* d_out, int out_size, void* d_ws, size_t ws_size,
                              hipStream_t stream) {
  (void)in_sizes; (void)n_in; (void)out_size; (void)d_ws; (void)ws_size;
  const float* qkv = (const float*)d_in[0];
  float* out = (float*)d_out;
  dim3 grid(SEQ / MT, NH, 2);   // (q-tiles, heads, batch)
  dim3 block(256);
  fa_fwd_f16wmma<<<grid, block, 0, stream>>>(qkv, out);
}
